// BoundaryFeaturePropagation_64673617543225
// MI455X (gfx1250) — matrix operand order repaired and checked by reading, not compiled, not run on hardware
//
#include <hip/hip_runtime.h>

#define DEVI __device__ __forceinline__

typedef __attribute__((ext_vector_type(4)))  unsigned int  u32x4;
typedef __attribute__((ext_vector_type(16))) __bf16        v16bf;
typedef __attribute__((ext_vector_type(8)))  float         v8f;
typedef __attribute__((ext_vector_type(4)))  float         f32x4;

union Frag32B { u32x4 q[2]; v16bf bf; };

// TDM descriptor vector types (this toolchain: 6-arg builtin,
// (uint32x4 g0, int32x8 g1, int32x4, int32x4, int32x8, i32 cpol))
typedef __attribute__((ext_vector_type(4))) unsigned int tdm_g0_t;
typedef __attribute__((ext_vector_type(8))) int          tdm_g1_t;
typedef __attribute__((ext_vector_type(4))) int          tdm_g2_t;
typedef __attribute__((ext_vector_type(8))) int          tdm_g4_t;

#if defined(__has_builtin)
#  if __has_builtin(__builtin_amdgcn_tensor_load_to_lds) && \
      __has_builtin(__builtin_amdgcn_s_wait_tensorcnt)
#    define USE_TDM 1
#  endif
#endif
#ifndef USE_TDM
#  define USE_TDM 0
#endif

// Problem dims (fixed by the reference)
constexpr int Bn  = 8;
constexpr int C   = 256;
constexpr int H   = 128;
constexpr int Wd  = 128;
constexpr int HW  = H * Wd;        // 16384
constexpr int CHW = C * HW;        // 4194304
constexpr int L   = 128;           // scan length
constexpr int APAD = 264;          // padded row stride (bf16 elems) for 16x256 LDS tiles

// ---- workspace layout (bytes) ----
constexpr size_t OFF_CONF  = 0;                       // B*H*W f32      = 524288
constexpr size_t OFF_PACK  = 524288;                  // 9*65536 bf16   = 1179648
constexpr size_t OFF_STATS = 1703936;                 // B*32*2 f32     = 2048
constexpr size_t OFF_ACCH  = 2097152;                 // B*C*H*W f32    = 134217728
constexpr size_t OFF_ACCV  = OFF_ACCH + 134217728;    // B*C*H*W f32
// total ~270.5 MB; y aliases accH (safe: per-tile read-before-write in outproj)

DEVI unsigned short f2bf(float f) {
  unsigned int u = __float_as_uint(f);
  u += 0x7FFFu + ((u >> 16) & 1u);         // round-to-nearest-even
  return (unsigned short)(u >> 16);
}

// B operand (32x16, bf16): lane L holds column n = L&15, kb = (L&16)?16:0,
// element e -> K = kbase + kb + e.  Pre-packed: each lane reads its own 32B.
DEVI v16bf load_bfrag(const unsigned short* base, int nt, int kc, int lane) {
  const unsigned short* p = base + ((((nt << 3) + kc) << 5) + lane) * 16;
  Frag32B f;
  f.q[0] = *(const u32x4*)(p);
  f.q[1] = *(const u32x4*)(p + 8);
  return f.bf;
}

// A operand (16x32, bf16) from a 16 x 256 (stride APAD) LDS tile:
// lane L: row m = L&15, kb = (L&16)?16:0; elems 0..15 -> K=kb+0..15
DEVI v16bf load_afrag(const unsigned short* buf, int kc, int lane) {
  int m  = lane & 15;
  int kb = (lane & 16) ? 16 : 0;
  const unsigned short* p = buf + m * APAD + kc * 32 + kb;
  Frag32B f;
  f.q[0] = *(const u32x4*)(p);
  f.q[1] = *(const u32x4*)(p + 8);
  return f.bf;
}

DEVI v8f wmma_bf16(v16bf a, v16bf b, v8f c) {
  return __builtin_amdgcn_wmma_f32_16x16x32_bf16(false, a, false, b, (short)0, c,
                                                 false, false);
}

// ---------------- Kernel 0: pack 9 CxC f32 weights -> bf16 B-fragment layout ----
// mat 0..3 = W_in[d], 4..7 = W_s[d], 8 = out_w.  B[k][n] = Wg[n*C + k].
__global__ __launch_bounds__(256) void pack_weights_kernel(
    const float* __restrict__ W_in, const float* __restrict__ W_s,
    const float* __restrict__ out_w, unsigned short* __restrict__ packed) {
  int idx  = blockIdx.x * 256 + threadIdx.x;       // 0 .. 9*65536-1
  int mat  = idx >> 16;
  int f    = idx & 65535;
  int e    = f & 15;
  int lane = (f >> 4) & 31;
  int kc   = (f >> 9) & 7;
  int nt   = (f >> 12) & 15;
  int n = nt * 16 + (lane & 15);
  int k = kc * 32 + ((lane & 16) ? 16 : 0) + e;
  const float* src = (mat < 4) ? (W_in + mat * 65536)
                   : (mat < 8) ? (W_s + (mat - 4) * 65536)
                               : out_w;
  packed[idx] = f2bf(src[n * C + k]);
}

// ---------------- Kernel 1: boundary confidence gate ----------------------------
__global__ __launch_bounds__(256) void conf_kernel(
    const float* __restrict__ bl, const float* __restrict__ beta,
    float* __restrict__ conf) {
  int i = blockIdx.x * 256 + threadIdx.x;          // B*H*W = 131072
  float x  = bl[i];
  float s1 = 1.0f / (1.0f + __expf(-x));
  float s2 = 1.0f / (1.0f + __expf(-(20.0f * s1 - 4.0f)));
  float c  = 1.0f - beta[0] * s2;
  conf[i]  = fminf(fmaxf(c, 0.0f), 1.0f);
}

// ---------------- Kernel 2: zero GN stats ---------------------------------------
__global__ __launch_bounds__(256) void zero_stats_kernel(float* __restrict__ stats) {
  stats[threadIdx.x]       = 0.0f;
  stats[256 + threadIdx.x] = 0.0f;
}

// ---------------- Kernel 3: directional recurrent scan (WMMA) -------------------
// blockIdx 0..63  -> dir = dirBase     (horizontal, sequences over W)
// blockIdx 64..127-> dir = dirBase + 2 (vertical,   sequences over H)
// 16 sequences per workgroup; 8 waves; wave w owns N-tiles {2w, 2w+1}.
// Double-buffered x/gate prefetch hides global latency behind the WMMA chain.
__global__ __launch_bounds__(256) void scan_kernel(
    const float* __restrict__ feature, const float* __restrict__ conf,
    const float* __restrict__ b_in, const float* __restrict__ b_s,
    const float* __restrict__ p_bias, const unsigned short* __restrict__ packed,
    float* __restrict__ accH, float* __restrict__ accV,
    int dirBase, int addMode) {
  __shared__ unsigned short WsL[65536];           // packed Ws fragments (128 KB)
  __shared__ unsigned short Axbuf[2][16 * APAD];  // x_l (bf16), double-buffered
  __shared__ unsigned short Hbuf[16 * APAD];      // g*h_{l-1} (bf16)
  __shared__ float addvec[256];                   // b_in + b_s + p_bias
  __shared__ float Gbuf[2][16];                   // gate at next position

  const int tid  = threadIdx.x;
  const int lane = tid & 31;
  const int wave = tid >> 5;
  const bool horiz = (blockIdx.x < 64);
  const int tile = blockIdx.x & 63;
  const int dir  = horiz ? dirBase : (dirBase + 2);
  const bool rev = (dir & 1);
  const int b    = tile >> 3;
  const int s0   = (tile & 7) * 16;               // h0 (horiz) or w0 (vert)
  float* outAcc  = horiz ? accH : accV;

  const unsigned short* wiP = packed + dir * 65536;
  const unsigned short* wsP = packed + (4 + dir) * 65536;

  // ---- Stage Ws fragments into LDS (reused for all 128 serial steps) ----
#if USE_TDM
  if (tid < 32) {  // one wave issues the Tensor Data Mover DMA (EXEC ignored)
    unsigned lds_off = (unsigned)(uintptr_t)(void*)WsL;  // LDS aperture: low 32 bits = byte offset
    unsigned long long ga = (unsigned long long)(uintptr_t)wsP;
    tdm_g0_t g0;
    g0[0] = 1u;                                   // count=1, user descriptor
    g0[1] = lds_off;                              // lds_addr
    g0[2] = (unsigned)(ga & 0xFFFFFFFFu);         // global_addr[31:0]
    g0[3] = (unsigned)((ga >> 32) & 0x01FFFFFFu)  // global_addr[56:32]
          | (2u << 30);                           // type = 2 ("image")
    tdm_g1_t g1;
    g1[0] = (int)(3u << 16);                      // data_size = 8B, no multicast
    g1[1] = (int)(16384u << 16);                  // tensor_dim0[15:0] = 16384
    g1[2] = (int)(1u << 16);                      // tensor_dim0 hi = 0, tensor_dim1 = 1
    g1[3] = (int)(16384u << 16);                  // tile_dim0 = 16384 (1-D, 128 KB)
    g1[4] = 0;                                    // tile_dim1/2 unused
    g1[5] = 16384;                                // tensor_dim0_stride
    g1[6] = 0; g1[7] = 0;
    tdm_g2_t gz = {0, 0, 0, 0};
    tdm_g4_t g4 = {0, 0, 0, 0, 0, 0, 0, 0};
    __builtin_amdgcn_tensor_load_to_lds(g0, g1, gz, gz, g4, 0);
    __builtin_amdgcn_s_wait_tensorcnt(0);
  }
#else
  for (int i = tid; i < 65536 / 8; i += 256)
    *(u32x4*)(WsL + i * 8) = *(const u32x4*)(wsP + i * 8);
#endif
  addvec[tid] = b_in[dir * 256 + tid] + b_s[dir * 256 + tid] + p_bias[dir * 256 + tid];
  for (int i = tid; i < 16 * APAD; i += 256) Hbuf[i] = 0;   // h_{-1} = 0

  const int sm  = tid & 15;                       // staging row
  const int scb = (tid >> 4) * 16;                // staging channel base

  // pre-stage x_0 -> Axbuf[0], g_1 -> Gbuf[0]
  {
    int pos0 = rev ? (L - 1) : 0;
    size_t gb = (size_t)b * CHW +
                (size_t)(horiz ? ((s0 + sm) * Wd + pos0) : (pos0 * Wd + s0 + sm));
    unsigned short tmp[16];
#pragma unroll
    for (int i = 0; i < 16; ++i)
      tmp[i] = f2bf(feature[gb + (size_t)(scb + i) * HW]);
    u32x4* dst = (u32x4*)(&Axbuf[0][sm * APAD + scb]);
    dst[0] = *(const u32x4*)(tmp);
    dst[1] = *(const u32x4*)(tmp + 8);
    if (tid < 16) {
      int p1 = rev ? (L - 2) : 1;
      Gbuf[0][tid] = conf[(size_t)b * HW +
                          (size_t)(horiz ? ((s0 + tid) * Wd + p1) : (p1 * Wd + s0 + tid))];
    }
  }
  __syncthreads();

  const int nt0 = wave * 2, nt1 = nt0 + 1;
  const int ncol0 = nt0 * 16 + (lane & 15);
  const int ncol1 = nt1 * 16 + (lane & 15);
  const int moff  = (lane & 16) ? 8 : 0;

  for (int l = 0; l < L; ++l) {
    const int cur = l & 1, nxt = cur ^ 1;
    const int pos = rev ? (L - 1 - l) : l;

    // ---- issue prefetch for x_{l+1} and g_{l+2} (overlaps with WMMA below) ----
    float xf[16];
    const bool havex = (l + 1 < L);
    if (havex) {
      int posn = rev ? (L - 2 - l) : (l + 1);
      size_t gb = (size_t)b * CHW +
                  (size_t)(horiz ? ((s0 + sm) * Wd + posn) : (posn * Wd + s0 + sm));
#pragma unroll
      for (int i = 0; i < 16; ++i)
        xf[i] = feature[gb + (size_t)(scb + i) * HW];
    }
    float gpre = 0.0f;
    if (tid < 16 && l + 2 < L) {
      int posg = rev ? (L - 3 - l) : (l + 2);
      gpre = conf[(size_t)b * HW +
                  (size_t)(horiz ? ((s0 + tid) * Wd + posg) : (posg * Wd + s0 + tid))];
    }

    // ---- compute: proj_l + (g*h_{l-1})*Ws^T ----
    v8f acc0 = {0.f, 0.f, 0.f, 0.f, 0.f, 0.f, 0.f, 0.f};
    v8f acc1 = {0.f, 0.f, 0.f, 0.f, 0.f, 0.f, 0.f, 0.f};
#pragma unroll
    for (int kc = 0; kc < 8; ++kc) {             // x_l * Wi^T (Wi frags live in VGPRs)
      v16bf a  = load_afrag(&Axbuf[cur][0], kc, lane);
      v16bf b0 = load_bfrag(wiP, nt0, kc, lane);
      v16bf b1 = load_bfrag(wiP, nt1, kc, lane);
      acc0 = wmma_bf16(a, b0, acc0);
      acc1 = wmma_bf16(a, b1, acc1);
    }
#pragma unroll
    for (int kc = 0; kc < 8; ++kc) {             // + (g*h_{l-1}) * Ws^T (LDS)
      v16bf a  = load_afrag(Hbuf, kc, lane);
      v16bf b0 = load_bfrag(WsL, nt0, kc, lane);
      v16bf b1 = load_bfrag(WsL, nt1, kc, lane);
      acc0 = wmma_bf16(a, b0, acc0);
      acc1 = wmma_bf16(a, b1, acc1);
    }

    // ---- commit prefetched x/gate into the other buffers ----
    if (havex) {
      unsigned short tmp[16];
#pragma unroll
      for (int i = 0; i < 16; ++i) tmp[i] = f2bf(xf[i]);
      u32x4* dst = (u32x4*)(&Axbuf[nxt][sm * APAD + scb]);
      dst[0] = *(const u32x4*)(tmp);
      dst[1] = *(const u32x4*)(tmp + 8);
    }
    if (tid < 16) Gbuf[nxt][tid] = gpre;
    __syncthreads();   // Hbuf reads + Axbuf[nxt]/Gbuf[nxt] writes complete

    // ---- epilogue: bias + ReLU, emit output, write gate-scaled state ----
    float av0 = addvec[ncol0], av1 = addvec[ncol1];
#pragma unroll
    for (int r = 0; r < 8; ++r) {
      int m = r + moff;
      float h0 = fmaxf(acc0[r] + av0, 0.0f);
      float h1 = fmaxf(acc1[r] + av1, 0.0f);
      size_t gb = (size_t)b * CHW +
                  (size_t)(horiz ? ((s0 + m) * Wd + pos) : (pos * Wd + s0 + m));
      size_t a0 = gb + (size_t)ncol0 * HW;
      size_t a1 = gb + (size_t)ncol1 * HW;
      if (addMode) { outAcc[a0] += h0; outAcc[a1] += h1; }
      else         { outAcc[a0]  = h0; outAcc[a1]  = h1; }
      float g = Gbuf[cur][m];
      Hbuf[m * APAD + ncol0] = f2bf(h0 * g);
      Hbuf[m * APAD + ncol1] = f2bf(h1 * g);
    }
    __syncthreads();   // epilogue writes visible before next step's reads
  }
}

// ---------------- Kernel 4: out_w GEMM + residual + GN partial stats ------------
__global__ __launch_bounds__(256) void outproj_kernel(
    const float* __restrict__ accH, const float* __restrict__ accV,
    const float* __restrict__ feature, const float* __restrict__ out_b,
    const unsigned short* __restrict__ packedOut,
    float* y /* aliases accH */, float* __restrict__ stats) {
  __shared__ unsigned short Xbuf[16 * APAD];
  __shared__ float sstat[64];                  // [32 groups][sum, sumsq]

  const int tid = threadIdx.x, lane = tid & 31, wave = tid >> 5;
  const int bimg = blockIdx.x >> 10;           // 1024 tiles / image
  const int p0   = (blockIdx.x & 1023) * 16;
  if (tid < 64) sstat[tid] = 0.0f;

  // stage fused = accH + accV (thread t = channel t, 16 contiguous points)
  {
    size_t gb = (size_t)bimg * CHW + (size_t)tid * HW + p0;
    unsigned short tmp[16];
#pragma unroll
    for (int j = 0; j < 16; ++j) tmp[j] = f2bf(accH[gb + j] + accV[gb + j]);
#pragma unroll
    for (int j = 0; j < 16; ++j) Xbuf[j * APAD + tid] = tmp[j];
  }
  __syncthreads();

  const int nt0 = wave * 2, nt1 = nt0 + 1;
  v8f acc0 = {0.f, 0.f, 0.f, 0.f, 0.f, 0.f, 0.f, 0.f};
  v8f acc1 = {0.f, 0.f, 0.f, 0.f, 0.f, 0.f, 0.f, 0.f};
#pragma unroll
  for (int kc = 0; kc < 8; ++kc) {
    v16bf a  = load_afrag(Xbuf, kc, lane);
    v16bf b0 = load_bfrag(packedOut, nt0, kc, lane);
    v16bf b1 = load_bfrag(packedOut, nt1, kc, lane);
    acc0 = wmma_bf16(a, b0, acc0);
    acc1 = wmma_bf16(a, b1, acc1);
  }

  const int n0 = nt0 * 16 + (lane & 15);
  const int n1 = n0 + 16;
  const int moff = (lane & 16) ? 8 : 0;
  float ob0 = out_b[n0], ob1 = out_b[n1];
  float s0 = 0.f, q0 = 0.f, s1 = 0.f, q1 = 0.f;
#pragma unroll
  for (int r = 0; r < 8; ++r) {
    int p = p0 + r + moff;
    size_t i0 = (size_t)bimg * CHW + (size_t)n0 * HW + p;
    size_t i1 = (size_t)bimg * CHW + (size_t)n1 * HW + p;
    float y0 = acc0[r] + ob0 + feature[i0];
    float y1 = acc1[r] + ob1 + feature[i1];
    y[i0] = y0; y[i1] = y1;
    s0 += y0; q0 += y0 * y0; s1 += y1; q1 += y1 * y1;
  }
  atomicAdd(&sstat[(n0 >> 3) * 2],     s0);
  atomicAdd(&sstat[(n0 >> 3) * 2 + 1], q0);
  atomicAdd(&sstat[(n1 >> 3) * 2],     s1);
  atomicAdd(&sstat[(n1 >> 3) * 2 + 1], q1);
  __syncthreads();
  if (tid < 64) atomicAdd(&stats[bimg * 64 + tid], sstat[tid]);
}

// ---------------- Kernel 5: GroupNorm finalize ----------------------------------
__global__ __launch_bounds__(256) void gn_kernel(
    const float* __restrict__ y, const float* __restrict__ stats,
    const float* __restrict__ gn_w, const float* __restrict__ gn_b,
    float* __restrict__ out) {
  size_t i4  = (size_t)blockIdx.x * 256 + threadIdx.x;  // float4 index
  size_t idx = i4 * 4;
  int bimg = (int)(idx / CHW);
  int c    = (int)((idx / HW) & (C - 1));
  int g    = c >> 3;
  const float cnt = 131072.0f;                 // (C/32)*H*W
  float s  = stats[bimg * 64 + g * 2];
  float q  = stats[bimg * 64 + g * 2 + 1];
  float mu = s / cnt;
  float var = q / cnt - mu * mu;
  float inv = rsqrtf(var + 1e-5f);
  float w = gn_w[c], bb = gn_b[c];
  f32x4 v = *(const f32x4*)(y + idx);
  f32x4 o;
#pragma unroll
  for (int k = 0; k < 4; ++k) o[k] = (v[k] - mu) * inv * w + bb;
  *(f32x4*)(out + idx) = o;
}

// ---------------- host launch ---------------------------------------------------
extern "C" void kernel_launch(void* const* d_in, const int* in_sizes, int n_in,
                              void* d_out, int out_size, void* d_ws, size_t ws_size,
                              hipStream_t stream) {
  const float* feature = (const float*)d_in[0];
  const float* blog    = (const float*)d_in[1];
  const float* beta    = (const float*)d_in[2];
  const float* W_in    = (const float*)d_in[3];
  const float* b_in    = (const float*)d_in[4];
  const float* W_s     = (const float*)d_in[5];
  const float* b_s     = (const float*)d_in[6];
  const float* p_bias  = (const float*)d_in[7];
  const float* out_w   = (const float*)d_in[8];
  const float* out_b   = (const float*)d_in[9];
  const float* gn_w    = (const float*)d_in[10];
  const float* gn_b    = (const float*)d_in[11];
  float* out = (float*)d_out;

  char* ws = (char*)d_ws;
  float*          conf   = (float*)(ws + OFF_CONF);
  unsigned short* packed = (unsigned short*)(ws + OFF_PACK);
  float*          stats  = (float*)(ws + OFF_STATS);
  float*          accH   = (float*)(ws + OFF_ACCH);
  float*          accV   = (float*)(ws + OFF_ACCV);

  pack_weights_kernel<<<9 * 65536 / 256, 256, 0, stream>>>(W_in, W_s, out_w, packed);
  conf_kernel<<<(Bn * HW) / 256, 256, 0, stream>>>(blog, beta, conf);
  zero_stats_kernel<<<1, 256, 0, stream>>>(stats);

  // phase A: lr (store accH) + tb (store accV)
  scan_kernel<<<128, 256, 0, stream>>>(feature, conf, b_in, b_s, p_bias, packed,
                                       accH, accV, /*dirBase=*/0, /*addMode=*/0);
  // phase B: rl (add accH) + bt (add accV)  — ordered by the stream
  scan_kernel<<<128, 256, 0, stream>>>(feature, conf, b_in, b_s, p_bias, packed,
                                       accH, accV, /*dirBase=*/1, /*addMode=*/1);

  // out projection + residual + GN partial stats  (y aliases accH)
  outproj_kernel<<<Bn * (HW / 16), 256, 0, stream>>>(accH, accV, feature, out_b,
                                                     packed + 8 * 65536, accH, stats);
  // GN finalize
  gn_kernel<<<(Bn * CHW) / 4 / 256, 256, 0, stream>>>(accH, stats, gn_w, gn_b, out);
}
